// MultiHeadAttention_3899830305178
// MI455X (gfx1250) — matrix operand order repaired and checked by reading, not compiled, not run on hardware
//
#include <hip/hip_runtime.h>
#include <math.h>

// ---------------------------------------------------------------------------
// MHA for MI455X (gfx1250, wave32, WMMA 16x16x32 f16 -> f32)
// B=2, S=2048, H=16, DK=64, DM=1024.  M = B*S = 4096.
// GEMMs: double-buffered LDS staging via global_load_async_to_lds_b128
// (ASYNCcnt pipeline), ds_load fragment reads, v_wmma f16->f32.
// Column softmax (reference softmax over axis=1 == query index) via online
// (max,sum) streaming pass, then recompute+normalize+LDS relayout+WMMA*V^T.
// ---------------------------------------------------------------------------

typedef __attribute__((ext_vector_type(16))) _Float16 v16h;
typedef __attribute__((ext_vector_type(8)))  _Float16 v8h;
typedef __attribute__((ext_vector_type(4)))  _Float16 v4h;
typedef __attribute__((ext_vector_type(8)))  float    v8f;

#define DEVINL static __device__ __forceinline__

DEVINL int lane_id() { return threadIdx.x & 31; }

DEVINL v16h cat8(v8h lo, v8h hi) {
    return __builtin_shufflevector(lo, hi, 0,1,2,3,4,5,6,7,8,9,10,11,12,13,14,15);
}

// ---------------- async global->LDS copy (CDNA5) ---------------------------
#if defined(__HIP_DEVICE_COMPILE__) && __has_builtin(__builtin_amdgcn_global_load_async_to_lds_b128)
#define HAVE_ASYNC_LDS 1
#else
#define HAVE_ASYNC_LDS 0
#endif

DEVINL void async_copy_b128(void* lds, const void* g) {
#if HAVE_ASYNC_LDS
    typedef int v4i __attribute__((vector_size(16)));
    __builtin_amdgcn_global_load_async_to_lds_b128(
        (__attribute__((address_space(1))) v4i*)g,
        (__attribute__((address_space(3))) v4i*)lds, 0, 0);
#else
    *(int4*)lds = *(const int4*)g;   // synchronous fallback
#endif
}

DEVINL void wait_async_le4() {
#if HAVE_ASYNC_LDS
#if __has_builtin(__builtin_amdgcn_s_wait_asynccnt)
    __builtin_amdgcn_s_wait_asynccnt(4);
#else
    asm volatile("s_wait_asynccnt 0x4" ::: "memory");
#endif
#endif
}

DEVINL void wait_async_0() {
#if HAVE_ASYNC_LDS
#if __has_builtin(__builtin_amdgcn_s_wait_asynccnt)
    __builtin_amdgcn_s_wait_asynccnt(0);
#else
    asm volatile("s_wait_asynccnt 0x0" ::: "memory");
#endif
#endif
}

// ---------------- WMMA fragment helpers ------------------------------------
// A fragment 16x32 f16 (MxK): lane: row = lane&15;
// halves[0..15] = contiguous K run (lane>>4)*16 .. +16
DEVINL v16h load_frag_a_f16(const _Float16* __restrict__ base, int ld) {
    int l = lane_id();
    const _Float16* p = base + (size_t)(l & 15) * ld + (l >> 4) * 16;
    return cat8(*(const v8h*)p, *(const v8h*)(p + 8));
}

// B fragment 32x16 (KxN) from the TRANSPOSED operand Bt[n][k]:
// lane: col n = lane&15; contiguous K run (lane>>4)*16 .. +16 (32B aligned).
DEVINL v16h load_frag_b(const _Float16* __restrict__ bt, int ld) {
    int l = lane_id();
    return *(const v16h*)(bt + (size_t)(l & 15) * ld + (l >> 4) * 16);
}

// Same fragments read from LDS tiles with pitch PITCH (16B-aligned v8h loads).
#define PITCH 40
DEVINL v16h lds_frag_a(const _Float16* base) {
    int l = lane_id();
    const _Float16* p = base + (l & 15) * PITCH + (l >> 4) * 16;
    return cat8(*(const v8h*)p, *(const v8h*)(p + 8));
}
DEVINL v16h lds_frag_b(const _Float16* base) {
    int l = lane_id();
    const _Float16* p = base + (l & 15) * PITCH + (l >> 4) * 16;
    return cat8(*(const v8h*)p, *(const v8h*)(p + 8));
}

DEVINL v8f wmma_f16(v16h a, v16h b, v8f c) {
    return __builtin_amdgcn_wmma_f32_16x16x32_f16(false, a, false, b, (short)0, c, false, false);
}

// ---------------------------------------------------------------------------
// Kernel: 1024x1024 fp32 -> f16 transpose (weights): Wt[n][k] = (f16)W[k][n]
// ---------------------------------------------------------------------------
__global__ void transpose_cvt_kernel(const float* __restrict__ in, _Float16* __restrict__ out) {
    __shared__ float tile[32][33];
    int x = blockIdx.x * 32 + threadIdx.x;
    int y = blockIdx.y * 32 + threadIdx.y;
#pragma unroll
    for (int t = 0; t < 32; t += 8)
        tile[threadIdx.y + t][threadIdx.x] = in[(size_t)(y + t) * 1024 + x];
    __syncthreads();
    int xo = blockIdx.y * 32 + threadIdx.x;
    int yo = blockIdx.x * 32 + threadIdx.y;
#pragma unroll
    for (int t = 0; t < 32; t += 8)
        out[(size_t)(yo + t) * 1024 + xo] = (_Float16)tile[threadIdx.x][threadIdx.y + t];
}

// ---------------------------------------------------------------------------
// Kernel: elementwise f32 -> f16 (activations), 4 elems/thread
// ---------------------------------------------------------------------------
__global__ void cvt_f16_kernel(const float* __restrict__ in, _Float16* __restrict__ out) {
    int i = blockIdx.x * blockDim.x + threadIdx.x;
    float4 x = ((const float4*)in)[i];
    v4h r;
    r[0] = (_Float16)x.x; r[1] = (_Float16)x.y;
    r[2] = (_Float16)x.z; r[3] = (_Float16)x.w;
    ((v4h*)out)[i] = r;
}

// ---------------------------------------------------------------------------
// GEMM: Out[4096 x 1024] = A[4096 x 1024] * Bt^T + bias   (A, Bt both f16)
//   MODE 0: f16 out row-major (q/k proj)   MODE 1: f16 out V-transposed
//   MODE 2: f32 out row-major (final projection)
// Block 256 = 8 waves; block tile 128x128; wave tile 64x32 (4x2 WMMA).
// Double-buffered LDS staging: 4 async b128 copies / thread / K-step.
// ---------------------------------------------------------------------------
template <int MODE>
__global__ void gemm_kernel(const _Float16* __restrict__ A,
                            const _Float16* __restrict__ Bt,
                            const float* __restrict__ bias,
                            void* __restrict__ Out) {
    __shared__ __align__(16) _Float16 As[2][128][PITCH];
    __shared__ __align__(16) _Float16 Bs[2][128][PITCH];
    const int lane = lane_id();
    const int wid  = threadIdx.x >> 5;
    const int wm = wid & 1, wn = wid >> 1;
    const int m0 = blockIdx.y * 128;
    const int n0 = blockIdx.x * 128;
    const int t  = threadIdx.x;

    // Stage one 128x32-half tile pair into buffer `buf`.
    // 512 16B chunks per tile; thread handles chunks t and t+256 of each.
    auto stage = [&](int buf, int k0) {
        for (int c = t; c < 512; c += 256) {
            int row = c >> 2, off = (c & 3) * 8;
            async_copy_b128(&As[buf][row][off], A  + (size_t)(m0 + row) * 1024 + k0 + off);
            async_copy_b128(&Bs[buf][row][off], Bt + (size_t)(n0 + row) * 1024 + k0 + off);
        }
    };

    stage(0, 0);
    v8f acc[4][2] = {};
    for (int k0 = 0, it = 0; k0 < 1024; k0 += 32, ++it) {
        const int buf = it & 1;
        if (k0 + 32 < 1024) { stage(buf ^ 1, k0 + 32); wait_async_le4(); }
        else                { wait_async_0(); }
        __syncthreads();

        v16h a[4], b[2];
#pragma unroll
        for (int mi = 0; mi < 4; ++mi) a[mi] = lds_frag_a(&As[buf][wm * 64 + mi * 16][0]);
#pragma unroll
        for (int ni = 0; ni < 2; ++ni) b[ni] = lds_frag_b(&Bs[buf][wn * 32 + ni * 16][0]);
#pragma unroll
        for (int mi = 0; mi < 4; ++mi)
#pragma unroll
            for (int ni = 0; ni < 2; ++ni)
                acc[mi][ni] = wmma_f16(a[mi], b[ni], acc[mi][ni]);
        __syncthreads();
    }

    const int col = lane & 15, rb = (lane >> 4) * 8;
#pragma unroll
    for (int mi = 0; mi < 4; ++mi)
#pragma unroll
        for (int ni = 0; ni < 2; ++ni)
#pragma unroll
            for (int r = 0; r < 8; ++r) {
                int row = m0 + wm * 64 + mi * 16 + rb + r;
                int cg  = n0 + wn * 32 + ni * 16 + col;
                float v = acc[mi][ni][r] + bias[cg];
                if constexpr (MODE == 0) {
                    ((_Float16*)Out)[(size_t)row * 1024 + cg] = (_Float16)v;
                } else if constexpr (MODE == 1) {
                    int bi = row >> 11, j = row & 2047;          // row = b*2048 + j
                    ((_Float16*)Out)[((size_t)(bi * 1024 + cg)) * 2048 + j] = (_Float16)v;
                } else {
                    ((float*)Out)[(size_t)row * 1024 + cg] = v;
                }
            }
}

// ---------------------------------------------------------------------------
// Column softmax stats: wave owns 16 columns j, streams all 2048 rows i.
// ---------------------------------------------------------------------------
__global__ void colstats_kernel(const _Float16* __restrict__ qp,
                                const _Float16* __restrict__ kp,
                                float* __restrict__ colmax,
                                float* __restrict__ colsum) {
    const int lane = lane_id();
    const int wid  = threadIdx.x >> 5;
    const int bh = blockIdx.y, b = bh >> 4, h = bh & 15;
    const int j0 = blockIdx.x * 128 + wid * 16;
    const float scale = 0.125f;

    const _Float16* kb = kp + ((size_t)(b * 2048 + j0)) * 1024 + h * 64;
    v16h b0 = load_frag_b(kb, 1024);
    v16h b1 = load_frag_b(kb + 32, 1024);

    float rmax = -INFINITY, rsum = 0.f;
    for (int i0 = 0; i0 < 2048; i0 += 16) {
        const _Float16* qb = qp + ((size_t)(b * 2048 + i0)) * 1024 + h * 64;
        v16h a0 = load_frag_a_f16(qb, 1024);
        v16h a1 = load_frag_a_f16(qb + 32, 1024);
        v8f s = {};
        s = wmma_f16(a0, b0, s);
        s = wmma_f16(a1, b1, s);
        float tmax = -INFINITY;
#pragma unroll
        for (int r = 0; r < 8; ++r) tmax = fmaxf(tmax, s[r] * scale);
        float nmax = fmaxf(rmax, tmax);
        float tsum = 0.f;
#pragma unroll
        for (int r = 0; r < 8; ++r) tsum += __expf(s[r] * scale - nmax);
        rsum = rsum * __expf(rmax - nmax) + tsum;
        rmax = nmax;
    }
    float omax = __shfl_xor(rmax, 16, 32);
    float osum = __shfl_xor(rsum, 16, 32);
    float m = fmaxf(rmax, omax);
    float sum = rsum * __expf(rmax - m) + osum * __expf(omax - m);
    if (lane < 16) {
        colmax[(size_t)bh * 2048 + j0 + lane] = m;
        colsum[(size_t)bh * 2048 + j0 + lane] = sum;
    }
}

// ---------------------------------------------------------------------------
// Attention output: wave owns 16 rows i, all 64 dims d of one (b,h); streams j.
// ---------------------------------------------------------------------------
__global__ void attn_kernel(const _Float16* __restrict__ qp,
                            const _Float16* __restrict__ kp,
                            const _Float16* __restrict__ vT,
                            const float* __restrict__ colmax,
                            const float* __restrict__ colsum,
                            _Float16* __restrict__ ao) {
    __shared__ __align__(64) _Float16 sh[8][16][32];
    const int lane = lane_id();
    const int wid  = threadIdx.x >> 5;
    const int bh = blockIdx.y, b = bh >> 4, h = bh & 15;
    const int i0 = blockIdx.x * 128 + wid * 16;
    const int col = lane & 15, rb8 = (lane >> 4) * 8;
    const float scale = 0.125f;

    const _Float16* qb = qp + ((size_t)(b * 2048 + i0)) * 1024 + h * 64;
    v16h aq0 = load_frag_a_f16(qb, 1024);
    v16h aq1 = load_frag_a_f16(qb + 32, 1024);

    v8f acc[4] = {};
    for (int j0 = 0; j0 < 2048; j0 += 32) {
#pragma unroll
        for (int t = 0; t < 2; ++t) {
            const int jb = j0 + t * 16;
            const _Float16* kb = kp + ((size_t)(b * 2048 + jb)) * 1024 + h * 64;
            v16h b0 = load_frag_b(kb, 1024);
            v16h b1 = load_frag_b(kb + 32, 1024);
            v8f s = {};
            s = wmma_f16(aq0, b0, s);
            s = wmma_f16(aq1, b1, s);
            const int j = jb + col;
            const float cm = colmax[(size_t)bh * 2048 + j];
            const float ci = 1.0f / colsum[(size_t)bh * 2048 + j];
#pragma unroll
            for (int r = 0; r < 8; ++r)
                sh[wid][rb8 + r][t * 16 + col] = (_Float16)(__expf(s[r] * scale - cm) * ci);
        }
        __syncthreads();
        v16h aw = load_frag_a_f16(&sh[wid][0][0], 32);
#pragma unroll
        for (int nf = 0; nf < 4; ++nf) {
            v16h bv = load_frag_b(vT + ((size_t)(bh * 64 + nf * 16)) * 2048 + j0, 2048);
            acc[nf] = wmma_f16(aw, bv, acc[nf]);
        }
        __syncthreads();
    }
#pragma unroll
    for (int nf = 0; nf < 4; ++nf)
#pragma unroll
        for (int r = 0; r < 8; ++r)
            ao[((size_t)(b * 2048 + i0 + rb8 + r)) * 1024 + h * 64 + nf * 16 + col] =
                (_Float16)acc[nf][r];
}

// ---------------------------------------------------------------------------
// Host launcher
// ---------------------------------------------------------------------------
extern "C" void kernel_launch(void* const* d_in, const int* in_sizes, int n_in,
                              void* d_out, int out_size, void* d_ws, size_t ws_size,
                              hipStream_t stream) {
    const float* query = (const float*)d_in[0];
    const float* key_i = (const float*)d_in[1];
    const float* value = (const float*)d_in[2];
    const float* Wq = (const float*)d_in[3];
    const float* bq = (const float*)d_in[4];
    const float* Wk = (const float*)d_in[5];
    const float* bk = (const float*)d_in[6];
    const float* Wv = (const float*)d_in[7];
    const float* bv = (const float*)d_in[8];
    const float* Wo = (const float*)d_in[9];
    const float* bo = (const float*)d_in[10];
    float* out = (float*)d_out;

    // Workspace (half elements, 1M = 1<<20):
    //  0..4M   : WqT,WkT,WvT,WoT       4M..16M : qp,kp,vT (projections)
    // 16M..28M : qf,kf,vf (f16 activations; ao aliases dead qf)
    // 28M..    : colmax/colsum (f32)                      total ~56.5 MB
    _Float16* wsh = (_Float16*)d_ws;
    _Float16* WqT = wsh + (size_t)0  * (1u << 20);
    _Float16* WkT = wsh + (size_t)1  * (1u << 20);
    _Float16* WvT = wsh + (size_t)2  * (1u << 20);
    _Float16* WoT = wsh + (size_t)3  * (1u << 20);
    _Float16* qp  = wsh + (size_t)4  * (1u << 20);
    _Float16* kp  = wsh + (size_t)8  * (1u << 20);
    _Float16* vT  = wsh + (size_t)12 * (1u << 20);
    _Float16* qf  = wsh + (size_t)16 * (1u << 20);
    _Float16* kf  = wsh + (size_t)20 * (1u << 20);
    _Float16* vf  = wsh + (size_t)24 * (1u << 20);
    _Float16* ao  = qf;                               // alias: qf dead after proj
    float* colmax = (float*)(wsh + (size_t)28 * (1u << 20));
    float* colsum = colmax + 32 * 2048;

    dim3 tb(32, 8), tg(32, 32);
    transpose_cvt_kernel<<<tg, tb, 0, stream>>>(Wq, WqT);
    transpose_cvt_kernel<<<tg, tb, 0, stream>>>(Wk, WkT);
    transpose_cvt_kernel<<<tg, tb, 0, stream>>>(Wv, WvT);
    transpose_cvt_kernel<<<tg, tb, 0, stream>>>(Wo, WoT);

    cvt_f16_kernel<<<4096, 256, 0, stream>>>(query, qf);
    cvt_f16_kernel<<<4096, 256, 0, stream>>>(key_i, kf);
    cvt_f16_kernel<<<4096, 256, 0, stream>>>(value, vf);

    dim3 gb(256), gg(8, 32);                       // N/128=8, M/128=32
    gemm_kernel<0><<<gg, gb, 0, stream>>>(qf, WqT, bq, qp);
    gemm_kernel<0><<<gg, gb, 0, stream>>>(kf, WkT, bk, kp);
    gemm_kernel<1><<<gg, gb, 0, stream>>>(vf, WvT, bv, vT);

    colstats_kernel<<<dim3(16, 32), 256, 0, stream>>>(qp, kp, colmax, colsum);
    attn_kernel<<<dim3(16, 32), 256, 0, stream>>>(qp, kp, vT, colmax, colsum, ao);

    gemm_kernel<2><<<gg, gb, 0, stream>>>(ao, WoT, bo, out);
}
